// Fusion_10720238370839
// MI455X (gfx1250) — compile-verified, matrix operand order checked by reading, not run on hardware
//
#include <hip/hip_runtime.h>

// CDNA5 / gfx1250. wave32. Main contraction runs on v_wmma_f32_16x16x32_bf16.
typedef __attribute__((ext_vector_type(16))) __bf16 v16bf;
typedef __attribute__((ext_vector_type(8)))  float  v8f;

#define D    128
#define HS   132        // padded LDS row stride (floats) -> conflict-free strided reads
#define TB   16         // tokens per workgroup (M tile)
#define NTOK 4096       // B*N
#define TAU  0.5f

__device__ __forceinline__ unsigned pack_bf16(float a, float b) {
  // round-to-nearest-even f32 -> bf16, packed pair (lo = a, hi = b)
  unsigned ua = __builtin_bit_cast(unsigned, a);
  unsigned ub = __builtin_bit_cast(unsigned, b);
  ua += 0x7FFFu + ((ua >> 16) & 1u);
  ub += 0x7FFFu + ((ub >> 16) & 1u);
  return (ua >> 16) | (ub & 0xFFFF0000u);
}

// Kernel 0: layerV f32 -> bf16 (2M elements, 8 per thread, 1024 blocks)
__global__ void cvt_layerV_bf16(const float4* __restrict__ src, uint4* __restrict__ dst) {
  size_t idx = (size_t)blockIdx.x * blockDim.x + threadIdx.x;
  float4 a = src[2 * idx], b = src[2 * idx + 1];
  uint4 o;
  o.x = pack_bf16(a.x, a.y); o.y = pack_bf16(a.z, a.w);
  o.z = pack_bf16(b.x, b.y); o.w = pack_bf16(b.z, b.w);
  dst[idx] = o;
}

union BU { uint4 q[2]; v16bf v; };   // one 32x16-slice B fragment chunk pair per jb
union AU { unsigned w[8]; v16bf v; };

// Kernel 1: fused  V[t,k] = sum_i f[t,i] * (h[t,:] @ layerV[k,i,:])  via bf16 WMMA,
//           then softmax(0.5*f_i*h_j) @ V. 256 WGs x 256 thr (8 wave32).
// WG = 16 tokens; wave w owns k-tile [16w, 16w+16).
__global__ __launch_bounds__(256)
void fused_bilinear_softmax(const float* __restrict__ hin,
                            const float* __restrict__ fin,
                            const unsigned short* __restrict__ lvbf,  // bf16 layerV [k][i][j]
                            float* __restrict__ out) {
  __shared__ float h_s[TB * HS];     // h, padded rows (softmax + A build)
  __shared__ float f_s[TB * HS];     // f, padded rows (softmax)
  __shared__ float f_t[D * TB];      // f transposed [i][t] (GEMM scale factors)
  __shared__ float V_s[TB * D];
  __shared__ float hmx[TB], hmn[TB];

  const int tid = threadIdx.x;
  const int t0  = blockIdx.x * TB;

  // ---- stage h, f (coalesced) ----
  for (int r = 0; r < 8; ++r) {
    int e = r * 256 + tid;              // 0..2047
    int t = e >> 7, d = e & 127;
    float hv = hin[(size_t)(t0 + t) * D + d];
    float fv = fin[(size_t)(t0 + t) * D + d];
    h_s[t * HS + d] = hv;
    f_s[t * HS + d] = fv;
    f_t[d * TB + t] = fv;               // transposed copy for the GEMM scale step
  }
  __syncthreads();

  // per-token h extrema for a stable softmax max: m = 0.5*f_i*(f_i>=0 ? hmax : hmin)
  if (tid < TB) {
    float mx = -1e30f, mn = 1e30f;
    for (int d2 = 0; d2 < D; ++d2) {
      float v = h_s[tid * HS + d2];
      mx = fmaxf(mx, v); mn = fminf(mn, v);
    }
    hmx[tid] = mx; hmn[tid] = mn;
  }

  const int lane  = tid & 31;
  const int wave  = tid >> 5;
  const int n     = lane & 15;   // A row (token) and B/D column (k) within tile
  const int hi    = lane >> 4;   // lane half selects K sub-chunks (16-bit WMMA layout)
  const int kbase = wave * 16;

  // ---- loop-invariant A fragments: A[jb] = h[t=n, j-chunks of jb] in bf16 ----
  AU Afr[4];
#pragma unroll
  for (int jb = 0; jb < 4; ++jb) {
    const int jlo = jb * 32 + hi * 8;   // words 0-3: K = 8*hi+0..7 ; words 4-7: +16
    const float4 hA = *(const float4*)&h_s[n * HS + jlo];
    const float4 hB = *(const float4*)&h_s[n * HS + jlo + 4];
    const float4 hC = *(const float4*)&h_s[n * HS + jlo + 16];
    const float4 hD = *(const float4*)&h_s[n * HS + jlo + 20];
    Afr[jb].w[0] = pack_bf16(hA.x, hA.y);
    Afr[jb].w[1] = pack_bf16(hA.z, hA.w);
    Afr[jb].w[2] = pack_bf16(hB.x, hB.y);
    Afr[jb].w[3] = pack_bf16(hB.z, hB.w);
    Afr[jb].w[4] = pack_bf16(hC.x, hC.y);
    Afr[jb].w[5] = pack_bf16(hC.z, hC.w);
    Afr[jb].w[6] = pack_bf16(hD.x, hD.y);
    Afr[jb].w[7] = pack_bf16(hD.z, hD.w);
  }

  // per-lane B base: row k = kbase+n of layerV, lane-half K sub-chunk offset
  const unsigned short* bbase = lvbf + (size_t)(kbase + n) * (D * D) + hi * 8;

  BU Bcur[4], Bnxt[4];
#pragma unroll
  for (int jb = 0; jb < 4; ++jb) {      // preload i = 0
    Bcur[jb].q[0] = *(const uint4*)(bbase + jb * 32);
    Bcur[jb].q[1] = *(const uint4*)(bbase + jb * 32 + 16);
  }

  v8f acc = {};
  for (int i = 0; i < D - 1; ++i) {
    const unsigned short* bnx = bbase + (size_t)(i + 1) * D;
    __builtin_prefetch(bnx + 8 * D, 0, 1);      // i+9 row -> global_prefetch_b8
#pragma unroll
    for (int jb = 0; jb < 4; ++jb) {            // software pipeline: fetch i+1
      Bnxt[jb].q[0] = *(const uint4*)(bnx + jb * 32);
      Bnxt[jb].q[1] = *(const uint4*)(bnx + jb * 32 + 16);
    }
    v8f P = {};
#pragma unroll
    for (int jb = 0; jb < 4; ++jb)              // P_i = h @ layerV[:,i,:]^T
      P = __builtin_amdgcn_wmma_f32_16x16x32_bf16(false, Afr[jb].v, false, Bcur[jb].v,
                                                  (short)0, P, false, false);
    // acc[t,k] += f[t,i] * P_i[t,k]; row t = r + 8*hi; two b128 LDS reads give 8 scales
    const float4 fA = *(const float4*)&f_t[i * TB + 8 * hi];
    const float4 fB = *(const float4*)&f_t[i * TB + 8 * hi + 4];
    acc[0] = fmaf(fA.x, P[0], acc[0]);
    acc[1] = fmaf(fA.y, P[1], acc[1]);
    acc[2] = fmaf(fA.z, P[2], acc[2]);
    acc[3] = fmaf(fA.w, P[3], acc[3]);
    acc[4] = fmaf(fB.x, P[4], acc[4]);
    acc[5] = fmaf(fB.y, P[5], acc[5]);
    acc[6] = fmaf(fB.z, P[6], acc[6]);
    acc[7] = fmaf(fB.w, P[7], acc[7]);
#pragma unroll
    for (int jb = 0; jb < 4; ++jb) Bcur[jb] = Bnxt[jb];
  }
  {                                             // peeled last iteration i = D-1
    v8f P = {};
#pragma unroll
    for (int jb = 0; jb < 4; ++jb)
      P = __builtin_amdgcn_wmma_f32_16x16x32_bf16(false, Afr[jb].v, false, Bcur[jb].v,
                                                  (short)0, P, false, false);
    const float4 fA = *(const float4*)&f_t[(D - 1) * TB + 8 * hi];
    const float4 fB = *(const float4*)&f_t[(D - 1) * TB + 8 * hi + 4];
    acc[0] = fmaf(fA.x, P[0], acc[0]);
    acc[1] = fmaf(fA.y, P[1], acc[1]);
    acc[2] = fmaf(fA.z, P[2], acc[2]);
    acc[3] = fmaf(fA.w, P[3], acc[3]);
    acc[4] = fmaf(fB.x, P[4], acc[4]);
    acc[5] = fmaf(fB.y, P[5], acc[5]);
    acc[6] = fmaf(fB.z, P[6], acc[6]);
    acc[7] = fmaf(fB.w, P[7], acc[7]);
  }

  // D layout (32-bit C/D 16x16): reg r, lane L -> M = r + 8*(L>>4), N = L&15
#pragma unroll
  for (int r = 0; r < 8; ++r)
    V_s[(r + 8 * hi) * D + kbase + n] = acc[r];
  __syncthreads();

  // ---- softmax over j of 0.5*f_i*h_j, weighted sum with V ----
  const int t  = tid >> 4;
  const int i0 = (tid & 15) * 8;
  for (int q = 0; q < 8; ++q) {
    int   ii = i0 + q;
    float fi = f_s[t * HS + ii];
    float m  = TAU * fi * (fi >= 0.f ? hmx[t] : hmn[t]);
    float s = 0.f, num = 0.f;
    for (int j = 0; j < D; ++j) {
      float e = __expf(TAU * fi * h_s[t * HS + j] - m);
      s   += e;
      num += e * V_s[t * D + j];
    }
    out[(size_t)(t0 + t) * D + ii] = num / s;
  }
}

extern "C" void kernel_launch(void* const* d_in, const int* in_sizes, int n_in,
                              void* d_out, int out_size, void* d_ws, size_t ws_size,
                              hipStream_t stream) {
  const float* h  = (const float*)d_in[0];
  const float* f  = (const float*)d_in[1];
  const float* lv = (const float*)d_in[2];
  float* out = (float*)d_out;

  // d_ws: 4 MB bf16 copy of layerV (2,097,152 elements)
  uint4* lvbf = (uint4*)d_ws;
  cvt_layerV_bf16<<<dim3((128 * 128 * 128) / (256 * 8)), dim3(256), 0, stream>>>(
      (const float4*)lv, lvbf);

  fused_bilinear_softmax<<<dim3(NTOK / TB), dim3(256), 0, stream>>>(
      h, f, (const unsigned short*)d_ws, out);
}
